// BothMamba_60713657696378
// MI455X (gfx1250) — compile-verified, matrix operand order checked by reading, not run on hardware
//
#include <hip/hip_runtime.h>

// ---------------------------------------------------------------------------
// BothMamba for gfx1250 (MI455X): WMMA f16 GEMMs + chunked parallel scan.
// ---------------------------------------------------------------------------

typedef __attribute__((ext_vector_type(16))) _Float16 v16h;
typedef __attribute__((ext_vector_type(4)))  _Float16 v4h;
typedef __attribute__((ext_vector_type(8)))  float    v8f;

#define NTOK   36864     // 4 * 9216
#define LSEQ   9216
#define NB     4
#define NCHUNK 32
#define CLEN   288       // 9216 / 32

__device__ __forceinline__ float silu_f(float v) { return v / (1.f + __expf(-v)); }
__device__ __forceinline__ float softplus_f(float t) {
  return (t > 20.f) ? t : __logf(1.f + __expf(t));
}

// ---------------------------------------------------------------------------
// 1) transpose: x (4,64,96,96) planar -> xs (4, 9216, 64) token-major
// ---------------------------------------------------------------------------
__global__ void k_transpose_bchw_blc(const float* __restrict__ x, float* __restrict__ xs) {
  size_t i = (size_t)blockIdx.x * 256 + threadIdx.x;
  if (i >= (size_t)NTOK * 64) return;
  size_t b = i / ((size_t)LSEQ * 64);
  size_t r = i % ((size_t)LSEQ * 64);
  size_t l = r / 64, c = r % 64;
  xs[i] = x[(b * 64 + c) * LSEQ + l];
}

// ---------------------------------------------------------------------------
// 2) WMMA GEMM:  C[M,N] = A[M,K] * W[N,K]^T   (f32 io, f16 compute)
//    K compile-time (64 or 128). block = 128 threads (4 waves);
//    block tile 16(M) x 64(N); entire K staged in LDS once, k-loop unrolled.
//    Requires M % 16 == 0 (true for all call sites: M = 36864).
// ---------------------------------------------------------------------------
template <int K>
__global__ void k_wmma_gemm(const float* __restrict__ A, const float* __restrict__ W,
                            float* __restrict__ C, int M, int N, int ldC) {
  __shared__ _Float16 sA[16 * K];   // [m][k]
  __shared__ _Float16 sB[64 * K];   // [n][k]
  const int tid  = threadIdx.x;
  const int wave = tid >> 5;
  const int lane = tid & 31;
  const int m0 = blockIdx.x * 16;
  const int n0 = blockIdx.y * 64;

  constexpr int KQ = K / 4;         // float4 quads per row

  // stage A tile (16 x K) as f16 : fixed-trip, vectorized
#pragma unroll
  for (int j = 0; j < (16 * KQ) / 128; ++j) {
    const int q  = tid + j * 128;
    const int m  = q / KQ, kq = q % KQ;
    const float4 va = *reinterpret_cast<const float4*>(A + (size_t)(m0 + m) * K + kq * 4);
    v4h hv = { (_Float16)va.x, (_Float16)va.y, (_Float16)va.z, (_Float16)va.w };
    *reinterpret_cast<v4h*>(&sA[m * K + kq * 4]) = hv;
  }
  // stage W tile (64 x K) as f16 (n-major rows of K == B^T); row-level N guard
#pragma unroll
  for (int j = 0; j < (64 * KQ) / 128; ++j) {
    const int q  = tid + j * 128;
    const int n  = q / KQ, kq = q % KQ;
    float4 vb = { 0.f, 0.f, 0.f, 0.f };
    if (n0 + n < N)
      vb = *reinterpret_cast<const float4*>(W + (size_t)(n0 + n) * K + kq * 4);
    v4h hv = { (_Float16)vb.x, (_Float16)vb.y, (_Float16)vb.z, (_Float16)vb.w };
    *reinterpret_cast<v4h*>(&sB[n * K + kq * 4]) = hv;
  }
  // speculative prefetch of the next M tile's A rows (global_prefetch_b8)
  if (m0 + 16 < M)
    __builtin_prefetch(A + (size_t)(m0 + 16 + (tid >> 3)) * K, 0, 0);
  __syncthreads();

  // wave32 fragments per ISA 7.12.2: A row = lane%16, K = (e<8?e:e+8)+8*(lane>=16)
  const int row  = lane & 15;
  const int koff = (lane >> 4) ? 8 : 0;
  const int nrow = wave * 16 + row;

  v8f acc = {};
#pragma unroll
  for (int kt = 0; kt < K / 32; ++kt) {
    v16h af, bf;
#pragma unroll
    for (int e = 0; e < 16; ++e) {
      const int kk = kt * 32 + (e < 8 ? e : e + 8) + koff;
      af[e] = sA[row * K + kk];
      bf[e] = sB[nrow * K + kk];
    }
    acc = __builtin_amdgcn_wmma_f32_16x16x32_f16(false, af, false, bf,
                                                 (short)0, acc, false, false);
  }

  // store C: lane<16 -> rows 0..7, lane>=16 -> rows 8..15; col = lane&15
  const int col = lane & 15;
  const int rb  = (lane >> 4) * 8;
  const int n   = n0 + wave * 16 + col;
  if (n < N) {
#pragma unroll
    for (int r = 0; r < 8; ++r)
      C[(size_t)(m0 + rb + r) * ldC + n] = acc[r];
  }
}

// ---------------------------------------------------------------------------
// 3) spatial depthwise causal conv(k=4) + bias + SiLU
//    xz (4,9216,256) cols [0,128) -> xc (4,9216,128)
// ---------------------------------------------------------------------------
__global__ void k_spa_conv_silu(const float* __restrict__ xz, const float* __restrict__ cw,
                                const float* __restrict__ cb, float* __restrict__ xc) {
  size_t i = (size_t)blockIdx.x * 256 + threadIdx.x;
  if (i >= (size_t)NTOK * 128) return;
  size_t t = i >> 7;              // global token
  int d = (int)(i & 127);
  int l = (int)(t % LSEQ);
  size_t base = (t - l) * 256;    // batch start * 256
  float s = cb[d];
#pragma unroll
  for (int j = 0; j < 4; ++j) {
    int ll = l - 3 + j;
    if (ll >= 0) s += cw[d * 4 + j] * xz[base + (size_t)ll * 256 + d];
  }
  xc[i] = silu_f(s);
}

// ---------------------------------------------------------------------------
// 4) chunked selective scan (3 phases). a[n] constant per (d,n) so the chunk
//    transition product is exp(a[n] * sum(dt)) -> only (h[16], sumdt) carried.
// ---------------------------------------------------------------------------
__global__ void k_scanA(const float* __restrict__ xdbl, const float* __restrict__ xc,
                        const float* __restrict__ dtw, const float* __restrict__ dtb,
                        const float* __restrict__ Alog,
                        float* __restrict__ hloc, float* __restrict__ sumdt) {
  const int b = blockIdx.x, ch = blockIdx.y, d = threadIdx.x;
  float a[16], h[16];
#pragma unroll
  for (int n = 0; n < 16; ++n) { a[n] = -__expf(Alog[d * 16 + n]); h[n] = 0.f; }
  const float w0 = dtw[d * 4 + 0], w1 = dtw[d * 4 + 1];
  const float w2 = dtw[d * 4 + 2], w3 = dtw[d * 4 + 3];
  const float bia = dtb[d];
  float sdt = 0.f;
  const int l0 = ch * CLEN;
  for (int i = 0; i < CLEN; ++i) {
    const size_t t = (size_t)b * LSEQ + l0 + i;
    const float* xr = xdbl + t * 36;
    float dt = softplus_f(w0 * xr[0] + w1 * xr[1] + w2 * xr[2] + w3 * xr[3] + bia);
    float xcv = xc[t * 128 + d];
    sdt += dt;
#pragma unroll
    for (int n = 0; n < 16; ++n)
      h[n] = __expf(dt * a[n]) * h[n] + dt * xr[4 + n] * xcv;
  }
  const size_t idx = (size_t)(b * NCHUNK + ch) * 128 + d;
#pragma unroll
  for (int n = 0; n < 16; ++n) hloc[idx * 16 + n] = h[n];
  sumdt[idx] = sdt;
}

__global__ void k_scanB(const float* __restrict__ hloc, const float* __restrict__ sumdt,
                        const float* __restrict__ Alog, float* __restrict__ carry) {
  const int b = blockIdx.x, d = threadIdx.x;
  float a[16], h[16];
#pragma unroll
  for (int n = 0; n < 16; ++n) { a[n] = -__expf(Alog[d * 16 + n]); h[n] = 0.f; }
  for (int c = 0; c < NCHUNK; ++c) {
    const size_t idx = (size_t)(b * NCHUNK + c) * 128 + d;
#pragma unroll
    for (int n = 0; n < 16; ++n) carry[idx * 16 + n] = h[n];
    const float sdt = sumdt[idx];
#pragma unroll
    for (int n = 0; n < 16; ++n)
      h[n] = __expf(a[n] * sdt) * h[n] + hloc[idx * 16 + n];
  }
}

__global__ void k_scanC(const float* __restrict__ xdbl, const float* __restrict__ xz,
                        const float* __restrict__ dtw, const float* __restrict__ dtb,
                        const float* __restrict__ Alog, const float* __restrict__ Dp,
                        const float* __restrict__ carry,
                        float* __restrict__ xc /* in: xc, out: y (in-place) */) {
  const int b = blockIdx.x, ch = blockIdx.y, d = threadIdx.x;
  float a[16], h[16];
  const size_t cidx = (size_t)(b * NCHUNK + ch) * 128 + d;
#pragma unroll
  for (int n = 0; n < 16; ++n) {
    a[n] = -__expf(Alog[d * 16 + n]);
    h[n] = carry[cidx * 16 + n];
  }
  const float w0 = dtw[d * 4 + 0], w1 = dtw[d * 4 + 1];
  const float w2 = dtw[d * 4 + 2], w3 = dtw[d * 4 + 3];
  const float bia = dtb[d], Dd = Dp[d];
  const int l0 = ch * CLEN;
  for (int i = 0; i < CLEN; ++i) {
    const size_t t = (size_t)b * LSEQ + l0 + i;
    const float* xr = xdbl + t * 36;
    float dt = softplus_f(w0 * xr[0] + w1 * xr[1] + w2 * xr[2] + w3 * xr[3] + bia);
    float xcv = xc[t * 128 + d];
    float accy = 0.f;
#pragma unroll
    for (int n = 0; n < 16; ++n) {
      h[n] = __expf(dt * a[n]) * h[n] + dt * xr[4 + n] * xcv;
      accy += h[n] * xr[20 + n];
    }
    float zv = xz[t * 256 + 128 + d];
    xc[t * 128 + d] = (accy + xcv * Dd) * silu_f(zv);
  }
}

// ---------------------------------------------------------------------------
// 5) GroupNorm(4 groups of 16ch over 96x96) + SiLU
//    in: token-major (4,9216,64); out: planar (4,64,9216)
// ---------------------------------------------------------------------------
__global__ void k_gn_silu(const float* __restrict__ yin, const float* __restrict__ g,
                          const float* __restrict__ bvec, float* __restrict__ out) {
  const int b = blockIdx.x >> 2, grp = blockIdx.x & 3;
  const int tid = threadIdx.x;
  __shared__ float s1[256], s2[256];
  float s = 0.f, q = 0.f;
  for (int cc = 0; cc < 16; ++cc) {
    const int c = grp * 16 + cc;
    for (int l = tid; l < LSEQ; l += 256) {
      float v = yin[((size_t)b * LSEQ + l) * 64 + c];
      s += v; q += v * v;
    }
  }
  s1[tid] = s; s2[tid] = q;
  __syncthreads();
  for (int st = 128; st > 0; st >>= 1) {
    if (tid < st) { s1[tid] += s1[tid + st]; s2[tid] += s2[tid + st]; }
    __syncthreads();
  }
  const float inv = 1.f / (16.f * LSEQ);
  const float mean = s1[0] * inv;
  const float var  = s2[0] * inv - mean * mean;
  const float rstd = rsqrtf(var + 1e-5f);
  for (int cc = 0; cc < 16; ++cc) {
    const int c = grp * 16 + cc;
    const float gg = g[c], bb = bvec[c];
    for (int l = tid; l < LSEQ; l += 256) {
      float v = yin[((size_t)b * LSEQ + l) * 64 + c];
      float u = (v - mean) * rstd * gg + bb;
      out[((size_t)(b * 64 + c)) * LSEQ + l] = silu_f(u);
    }
  }
}

// ---------------------------------------------------------------------------
// 6) spectral Mamba: per pixel, L=8 tokens of 8 ch; 16 threads/pixel (d<16),
//    16 pixels / 256-thread block; all intermediates in LDS.
//    out: token-major (4,9216,64), channel c = t*8 + dm
// ---------------------------------------------------------------------------
__global__ void k_spe_mamba(const float* __restrict__ x, const float* __restrict__ in_w,
                            const float* __restrict__ cw, const float* __restrict__ cb,
                            const float* __restrict__ xw, const float* __restrict__ dtw,
                            const float* __restrict__ dtb, const float* __restrict__ Alog,
                            const float* __restrict__ Dp, const float* __restrict__ ow,
                            float* __restrict__ out) {
  __shared__ float s_xt[16][8][8];
  __shared__ float s_xc[16][8][16];
  __shared__ float s_B[16][8][16];
  __shared__ float s_C[16][8][16];
  __shared__ float s_dt[16][8];
  __shared__ float s_y[16][8][16];
  const int tid = threadIdx.x;
  const int pix = tid >> 4, d = tid & 15;
  const size_t P = (size_t)blockIdx.x * 16 + pix;
  const size_t b = P / LSEQ, l = P % LSEQ;
  const float* xb = x + (b * 64) * LSEQ + l;

  for (int i = d; i < 64; i += 16)
    s_xt[pix][i >> 3][i & 7] = xb[(size_t)i * LSEQ];
  __syncthreads();

  // in-proj (ch d of 16) + z, causal conv(k=4) + bias + SiLU
  float pre[8], zv[8], xcl[8];
#pragma unroll
  for (int t = 0; t < 8; ++t) {
    float sxc = 0.f, sz = 0.f;
#pragma unroll
    for (int g = 0; g < 8; ++g) {
      float v = s_xt[pix][t][g];
      sxc += v * in_w[d * 8 + g];
      sz  += v * in_w[(16 + d) * 8 + g];
    }
    pre[t] = sxc; zv[t] = sz;
  }
#pragma unroll
  for (int t = 0; t < 8; ++t) {
    float s = cb[d];
#pragma unroll
    for (int j = 0; j < 4; ++j) {
      int tt = t - 3 + j;
      if (tt >= 0) s += cw[d * 4 + j] * pre[tt];
    }
    xcl[t] = silu_f(s);
    s_xc[pix][t][d] = xcl[t];
  }
  __syncthreads();

  // x-proj: B rows 1..16, C rows 17..32, dt raw row 0
#pragma unroll
  for (int t = 0; t < 8; ++t) {
    float sb = 0.f, sc = 0.f;
#pragma unroll
    for (int k = 0; k < 16; ++k) {
      float v = s_xc[pix][t][k];
      sb += v * xw[(1 + d) * 16 + k];
      sc += v * xw[(17 + d) * 16 + k];
    }
    s_B[pix][t][d] = sb; s_C[pix][t][d] = sc;
    if (d == 0) {
      float sd = 0.f;
#pragma unroll
      for (int k = 0; k < 16; ++k) sd += s_xc[pix][t][k] * xw[k];
      s_dt[pix][t] = sd;
    }
  }
  __syncthreads();

  // scan (L=8)
  float a[16], h[16];
#pragma unroll
  for (int n = 0; n < 16; ++n) { a[n] = -__expf(Alog[d * 16 + n]); h[n] = 0.f; }
  const float dtwd = dtw[d], dtbd = dtb[d], Dd = Dp[d];
#pragma unroll
  for (int t = 0; t < 8; ++t) {
    float dt = softplus_f(s_dt[pix][t] * dtwd + dtbd);
    float xcv = xcl[t];
    float acc = 0.f;
#pragma unroll
    for (int n = 0; n < 16; ++n) {
      h[n] = __expf(dt * a[n]) * h[n] + dt * s_B[pix][t][n] * xcv;
      acc += h[n] * s_C[pix][t][n];
    }
    s_y[pix][t][d] = (acc + xcv * Dd) * silu_f(zv[t]);
  }
  __syncthreads();

  // out-proj: 8 tokens x 8 out-ch per pixel
  for (int i = d; i < 64; i += 16) {
    int t = i >> 3, dm = i & 7;
    float s = 0.f;
#pragma unroll
    for (int k = 0; k < 16; ++k) s += s_y[pix][t][k] * ow[dm * 16 + k];
    out[(b * LSEQ + l) * 64 + (t * 8 + dm)] = s;
  }
}

// ---------------------------------------------------------------------------
// 7) fusion: out = w0*spaGN + w1*speGN + x   (all planar)
// ---------------------------------------------------------------------------
__global__ void k_fuse(const float* __restrict__ spa, const float* __restrict__ spe,
                       const float* __restrict__ x, const float* __restrict__ fw,
                       float* __restrict__ out) {
  size_t i = (size_t)blockIdx.x * 256 + threadIdx.x;
  if (i >= (size_t)NTOK * 64) return;
  float e0 = __expf(fw[0]), e1 = __expf(fw[1]);
  float w0 = e0 / (e0 + e1);
  out[i] = spa[i] * w0 + spe[i] * (1.f - w0) + x[i];
}

// ---------------------------------------------------------------------------
// host
// ---------------------------------------------------------------------------
extern "C" void kernel_launch(void* const* d_in, const int* in_sizes, int n_in,
                              void* d_out, int out_size, void* d_ws, size_t ws_size,
                              hipStream_t stream) {
  const float* x        = (const float*)d_in[0];
  const float* spa_in_w = (const float*)d_in[1];
  const float* spa_cw   = (const float*)d_in[2];
  const float* spa_cb   = (const float*)d_in[3];
  const float* spa_xw   = (const float*)d_in[4];
  const float* spa_dtw  = (const float*)d_in[5];
  const float* spa_dtb  = (const float*)d_in[6];
  const float* spa_Alog = (const float*)d_in[7];
  const float* spa_D    = (const float*)d_in[8];
  const float* spa_ow   = (const float*)d_in[9];
  const float* spa_g    = (const float*)d_in[10];
  const float* spa_b    = (const float*)d_in[11];
  const float* spe_in_w = (const float*)d_in[12];
  const float* spe_cw   = (const float*)d_in[13];
  const float* spe_cb   = (const float*)d_in[14];
  const float* spe_xw   = (const float*)d_in[15];
  const float* spe_dtw  = (const float*)d_in[16];
  const float* spe_dtb  = (const float*)d_in[17];
  const float* spe_Alog = (const float*)d_in[18];
  const float* spe_D    = (const float*)d_in[19];
  const float* spe_ow   = (const float*)d_in[20];
  const float* spe_g    = (const float*)d_in[21];
  const float* spe_b    = (const float*)d_in[22];
  const float* fuse_w   = (const float*)d_in[23];
  float* out = (float*)d_out;

  // workspace layout (floats); total ~18.4M floats (~73.5 MB)
  float* ws    = (float*)d_ws;
  float* xs    = ws;                      // 2,359,296  (xs; reused for spe token-major)
  float* xz    = xs   + 2359296;          // 9,437,184  (xz; reused: ys / spaGN / speGN)
  float* xc    = xz   + 9437184;          // 4,718,592  (xc; reused in-place for y)
  float* xdbl  = xc   + 4718592;          // 1,327,104
  float* hloc  = xdbl + 1327104;          //   262,144
  float* sumdt = hloc + 262144;           //    16,384
  float* carry = sumdt + 16384;           //   262,144
  float* ys      = xz;                    // spa out-proj result, token-major
  float* spa_gn  = xz + 2359296;          // planar
  float* spe_gn  = xz + 4718592;          // planar
  float* spe_tok = xs;                    // spe result, token-major

  // 1) x -> xs (token-major)
  k_transpose_bchw_blc<<<9216, 256, 0, stream>>>(x, xs);

  // 2) spatial in-proj: (36864x64)*(256x64)^T -> xz (ld 256)   [WMMA K=64]
  k_wmma_gemm<64><<<dim3(NTOK / 16, 4), 128, 0, stream>>>(xs, spa_in_w, xz,
                                                          NTOK, 256, 256);
  // 3) depthwise conv + SiLU -> xc
  k_spa_conv_silu<<<(NTOK * 128) / 256, 256, 0, stream>>>(xz, spa_cw, spa_cb, xc);

  // 4) x-proj: (36864x128)*(36x128)^T -> xdbl (ld 36)          [WMMA K=128]
  k_wmma_gemm<128><<<dim3(NTOK / 16, 1), 128, 0, stream>>>(xc, spa_xw, xdbl,
                                                           NTOK, 36, 36);

  // 5) chunked selective scan (y overwrites xc)
  k_scanA<<<dim3(NB, NCHUNK), 128, 0, stream>>>(xdbl, xc, spa_dtw, spa_dtb,
                                                spa_Alog, hloc, sumdt);
  k_scanB<<<NB, 128, 0, stream>>>(hloc, sumdt, spa_Alog, carry);
  k_scanC<<<dim3(NB, NCHUNK), 128, 0, stream>>>(xdbl, xz, spa_dtw, spa_dtb,
                                                spa_Alog, spa_D, carry, xc);

  // 6) out-proj: (36864x128)*(64x128)^T -> ys (ld 64)          [WMMA K=128]
  k_wmma_gemm<128><<<dim3(NTOK / 16, 1), 128, 0, stream>>>(xc, spa_ow, ys,
                                                           NTOK, 64, 64);
  // 7) GN+SiLU spatial
  k_gn_silu<<<16, 256, 0, stream>>>(ys, spa_g, spa_b, spa_gn);

  // 8) spectral Mamba -> spe_tok (token-major)
  k_spe_mamba<<<NTOK / 16, 256, 0, stream>>>(x, spe_in_w, spe_cw, spe_cb, spe_xw,
                                             spe_dtw, spe_dtb, spe_Alog, spe_D,
                                             spe_ow, spe_tok);
  // 9) GN+SiLU spectral
  k_gn_silu<<<16, 256, 0, stream>>>(spe_tok, spe_g, spe_b, spe_gn);

  // 10) fuse + residual
  k_fuse<<<9216, 256, 0, stream>>>(spa_gn, spe_gn, x, fuse_w, out);
}
